// Spectrogram_generator_24154896073011
// MI455X (gfx1250) — matrix operand order repaired and checked by reading, not compiled, not run on hardware
//
#include <hip/hip_runtime.h>
#include <hip/hip_bf16.h>

typedef _Float16 half_t;
typedef __attribute__((ext_vector_type(16))) _Float16 v16h;
typedef __attribute__((ext_vector_type(8)))  float    v8f;

#define BB 64
#define CC 513
#define TT 257
#define HH 256
#define KK 8192
#define NROWS (BB*TT)       /* 16448 = 1028 row-tiles of 16 */
#define XTS 520             /* padded K stride (multiple of 8, >=513) */
#define LDSP 264            /* LDS row stride in halves (bank-conflict free) */
#define LN_EPS 1e-5f
#define LEAK 0.01f

// ---------------- fragment loaders (global memory) ----------------
// A-frag 16x32 f16: lane holds row m=lane&15; half h=lane>>4 covers
// K = {h*8..h*8+7} U {16+h*8..16+h*8+7}  -> two b128 loads.
__device__ __forceinline__ v16h ldA_g(const half_t* base, int row, int rstride, int k0) {
  int lane = threadIdx.x & 31;
  int m = lane & 15, hf = lane >> 4;
  const half_t* q = base + (size_t)(row + m) * rstride + (k0 + hf * 8);
  v16h r;
  reinterpret_cast<float4*>(&r)[0] = *reinterpret_cast<const float4*>(q);
  reinterpret_cast<float4*>(&r)[1] = *reinterpret_cast<const float4*>(q + 16);
  return r;
}
// B-frag 32x16 f16: lane holds column n=lane&15; half h covers K = {h*8..h*8+7} U {16+h*8..16+h*8+7}
__device__ __forceinline__ v16h ldB_g(const half_t* base, int col, int rstride, int k0) {
  int lane = threadIdx.x & 31;
  int n = lane & 15, hf = lane >> 4;
  const half_t* q = base + (size_t)(col + n) * rstride + (k0 + hf * 8);
  v16h r;
  reinterpret_cast<float4*>(&r)[0] = *reinterpret_cast<const float4*>(q);
  reinterpret_cast<float4*>(&r)[1] = *reinterpret_cast<const float4*>(q + 16);
  return r;
}

__device__ __forceinline__ float red16(float v) {  // sum across 16-lane half
  v += __shfl_xor(v, 1, 32);
  v += __shfl_xor(v, 2, 32);
  v += __shfl_xor(v, 4, 32);
  v += __shfl_xor(v, 8, 32);
  return v;
}

// CDNA5 async global->LDS copy (ASYNCcnt-tracked DMA, §15.18 VGLOBAL).
// INST_OFFSET is added to BOTH the LDS and the global address, so one base
// register pair + immediate offsets covers a contiguous per-thread copy.
template <int OFS>
__device__ __forceinline__ void async_b128(unsigned lds_off, const void* g) {
  asm volatile("global_load_async_to_lds_b128 %0, %1, off offset:%2"
               :: "v"(lds_off), "v"(g), "n"(OFS) : "memory");
}
__device__ __forceinline__ void async_copy_256B(unsigned lds_off, const void* g) {
  async_b128<0>(lds_off, g);   async_b128<16>(lds_off, g);
  async_b128<32>(lds_off, g);  async_b128<48>(lds_off, g);
  async_b128<64>(lds_off, g);  async_b128<80>(lds_off, g);
  async_b128<96>(lds_off, g);  async_b128<112>(lds_off, g);
  async_b128<128>(lds_off, g); async_b128<144>(lds_off, g);
  async_b128<160>(lds_off, g); async_b128<176>(lds_off, g);
  async_b128<192>(lds_off, g); async_b128<208>(lds_off, g);
  async_b128<224>(lds_off, g); async_b128<240>(lds_off, g);
}
__device__ __forceinline__ void wait_async0() {
  asm volatile("s_wait_asynccnt 0x0" ::: "memory");
}

// ---------------- prep kernels ----------------
__global__ __launch_bounds__(256) void k_xt(const float* __restrict__ inp, half_t* __restrict__ xt) {
  __shared__ float tile[32 * 33];
  int b = blockIdx.z;
  int t0 = blockIdx.x * 32, c0 = blockIdx.y * 32;
  int tx = threadIdx.x, ty = threadIdx.y;  // 32 x 8
#pragma unroll
  for (int i = 0; i < 4; ++i) {
    int cl = ty + i * 8;
    int c = c0 + cl, t = t0 + tx;
    float v = 0.f;
    if (c < CC && t < TT) v = inp[((size_t)b * CC + c) * TT + t];
    tile[cl * 33 + tx] = v;
  }
  __syncthreads();
#pragma unroll
  for (int i = 0; i < 4; ++i) {
    int tl = ty + i * 8;
    int t = t0 + tl, c = c0 + tx;
    if (t < TT && c < XTS)
      xt[((size_t)b * TT + t) * XTS + c] = (half_t)tile[tx * 33 + tl];
  }
}

__global__ __launch_bounds__(256) void k_w1t(const float* __restrict__ W1, half_t* __restrict__ w1t) {
  int n = blockIdx.x;
  for (int k = threadIdx.x; k < XTS; k += 256)
    w1t[(size_t)n * XTS + k] = (k < CC) ? (half_t)W1[(size_t)k * HH + n] : (half_t)0.f;
}

__global__ __launch_bounds__(256) void k_w2t(const float* __restrict__ W2, half_t* __restrict__ w2t) {
  int n = blockIdx.x, k = threadIdx.x;
  w2t[(size_t)n * HH + k] = (half_t)W2[(size_t)k * HH + n];
}

__global__ __launch_bounds__(256) void k_embed(const float* __restrict__ emb, half_t* __restrict__ eh,
                                               float* __restrict__ enorm) {
  int row = blockIdx.x * 16 + (threadIdx.x >> 4);
  int j0 = (threadIdx.x & 15) * 16;
  const float* src = emb + (size_t)row * HH + j0;
  half_t* dst = eh + (size_t)row * HH + j0;
  float s = 0.f;
#pragma unroll
  for (int j = 0; j < 16; ++j) { float v = src[j]; s += v * v; dst[j] = (half_t)v; }
  s = red16(s);
  if ((threadIdx.x & 15) == 0) enorm[row] = s;
}

// ---------------- fused GEMM1 + LN + leaky + GEMM2 ----------------
// block = 4 waves; each wave owns one 16-row tile. grid = 257 (1028 tiles).
// K-outer / tile-inner loop order: A-frag loaded once per K-step, 16
// independent accumulator chains -> no serial WMMA RAW, partial load waits.
__global__ __launch_bounds__(128) void k_mlp(const half_t* __restrict__ xt, const half_t* __restrict__ w1t,
                                             const half_t* __restrict__ w2t,
                                             const float* __restrict__ b1, const float* __restrict__ gamma,
                                             const float* __restrict__ beta, const float* __restrict__ b2,
                                             float* __restrict__ v32, half_t* __restrict__ vh,
                                             float* __restrict__ vnorm) {
  __shared__ half_t hsh[4 * 16 * LDSP];
  int wave = threadIdx.x >> 5, lane = threadIdx.x & 31;
  int n16 = lane & 15, hf = lane >> 4;
  int rowBase = (blockIdx.x * 4 + wave) * 16;

  v8f acc[16];
#pragma unroll
  for (int t = 0; t < 16; ++t) acc[t] = (v8f){};

  // GEMM1: K = 0..511 via WMMA
  for (int ks = 0; ks < 16; ++ks) {
    v16h a = ldA_g(xt, rowBase, XTS, ks * 32);
#pragma unroll
    for (int t = 0; t < 16; ++t) {
      v16h b = ldB_g(w1t, t * 16, XTS, ks * 32);
      acc[t] = __builtin_amdgcn_wmma_f32_16x16x32_f16(false, a, false, b, (short)0, acc[t], false, false);
    }
  }
  // rank-1 tail for K=512 (K=513 is odd) + bias
  float xlast[8];
#pragma unroll
  for (int r = 0; r < 8; ++r)
    xlast[r] = (float)xt[(size_t)(rowBase + hf * 8 + r) * XTS + 512];
#pragma unroll
  for (int t = 0; t < 16; ++t) {
    int n = t * 16 + n16;
    float wl = (float)w1t[(size_t)n * XTS + 512];
    float bias = b1[n];
#pragma unroll
    for (int r = 0; r < 8; ++r) acc[t][r] += xlast[r] * wl + bias;
  }

  // LayerNorm over H=256: row m = r + 8*hf lives on the 16 lanes of one half
  float mu[8], inv[8];
  {
    float s1[8], s2[8];
#pragma unroll
    for (int r = 0; r < 8; ++r) { s1[r] = 0.f; s2[r] = 0.f; }
#pragma unroll
    for (int t = 0; t < 16; ++t)
#pragma unroll
      for (int r = 0; r < 8; ++r) { float h = acc[t][r]; s1[r] += h; s2[r] += h * h; }
#pragma unroll
    for (int r = 0; r < 8; ++r) {
      float a = red16(s1[r]) * (1.f / 256.f);
      float b = red16(s2[r]) * (1.f / 256.f);
      mu[r] = a;
      inv[r] = rsqrtf(b - a * a + LN_EPS);
    }
  }

  half_t* myh = &hsh[wave * 16 * LDSP];
#pragma unroll
  for (int t = 0; t < 16; ++t) {
    int n = t * 16 + n16;
    float g = gamma[n], bt = beta[n];
#pragma unroll
    for (int r = 0; r < 8; ++r) {
      float h = (acc[t][r] - mu[r]) * inv[r] * g + bt;
      h = (h >= 0.f) ? h : LEAK * h;
      myh[(hf * 8 + r) * LDSP + n] = (half_t)h;
    }
  }
  __syncthreads();

  // GEMM2: v = h1 @ W2 + b2 ; A from LDS (conflict-free stride 264)
#pragma unroll
  for (int t = 0; t < 16; ++t) acc[t] = (v8f){};
  for (int ks = 0; ks < 8; ++ks) {
    v16h a;
    {
      const half_t* q = &myh[(lane & 15) * LDSP + ks * 32 + hf * 8];
      reinterpret_cast<float4*>(&a)[0] = *reinterpret_cast<const float4*>(q);
      reinterpret_cast<float4*>(&a)[1] = *reinterpret_cast<const float4*>(q + 16);
    }
#pragma unroll
    for (int t = 0; t < 16; ++t) {
      v16h b = ldB_g(w2t, t * 16, HH, ks * 32);
      acc[t] = __builtin_amdgcn_wmma_f32_16x16x32_f16(false, a, false, b, (short)0, acc[t], false, false);
    }
  }
  float vn[8];
#pragma unroll
  for (int r = 0; r < 8; ++r) vn[r] = 0.f;
#pragma unroll
  for (int t = 0; t < 16; ++t) {
    int n = t * 16 + n16;
    float bias = b2[n];
#pragma unroll
    for (int r = 0; r < 8; ++r) {
      float v = acc[t][r] + bias;
      size_t off = (size_t)(rowBase + hf * 8 + r) * HH + n;
      v32[off] = v;
      vh[off] = (half_t)v;
      vn[r] += v * v;
    }
  }
#pragma unroll
  for (int r = 0; r < 8; ++r) vn[r] = red16(vn[r]);
  if (n16 == 0) {
#pragma unroll
    for (int r = 0; r < 8; ++r) vnorm[rowBase + hf * 8 + r] = vn[r];
  }
}

// ---------------- VQ nearest-neighbor search (the 69 GF GEMM) ----------------
// 64-embedding tiles, double-buffered LDS filled by async global->LDS DMA:
// issue tile i+1 into the other buffer, compute 32 WMMAs on tile i while the
// DMA runs. One s_wait_asynccnt + one barrier per iteration.
__global__ __launch_bounds__(128) void k_vq(const half_t* __restrict__ vh, const half_t* __restrict__ eh,
                                            const float* __restrict__ vnorm, const float* __restrict__ enorm,
                                            int* __restrict__ idx) {
  __shared__ half_t esh[2][64 * LDSP];
  int wave = threadIdx.x >> 5, lane = threadIdx.x & 31;
  int n16 = lane & 15, hf = lane >> 4;
  int rowBase = (blockIdx.x * 4 + wave) * 16;

  v16h af[8];  // entire 16x256 v tile resident in VGPRs
#pragma unroll
  for (int ks = 0; ks < 8; ++ks) af[ks] = ldA_g(vh, rowBase, HH, ks * 32);

  float vn[8];
#pragma unroll
  for (int r = 0; r < 8; ++r) vn[r] = vnorm[rowBase + hf * 8 + r];

  float bd[8]; int be[8];
#pragma unroll
  for (int r = 0; r < 8; ++r) { bd[r] = 3.4e38f; be[r] = 0; }

  // per-thread DMA slice: half an embedding row (128 halves = 256 B)
  int crow = threadIdx.x >> 1;         // 0..63
  int cseg = threadIdx.x & 1;          // 0..1
  unsigned ldsSlice = (unsigned)(size_t)&esh[0][crow * LDSP + cseg * 128];
  unsigned bufStride = (unsigned)((size_t)&esh[1][0] - (size_t)&esh[0][0]);
  const half_t* gslice = eh + (size_t)crow * HH + cseg * 128;

  async_copy_256B(ldsSlice, gslice);   // prologue: tile 0 -> buf 0

  for (int it = 0; it < KK / 64; ++it) {
    wait_async0();
    __syncthreads();   // tile `it` resident; everyone done reading buf[(it+1)&1]
    if (it + 1 < KK / 64)
      async_copy_256B(ldsSlice + ((it + 1) & 1) * bufStride,
                      gslice + (size_t)(it + 1) * 64 * HH);

    const half_t* cur = &esh[it & 1][0];
#pragma unroll
    for (int j = 0; j < 4; ++j) {
      v8f c = {};
#pragma unroll
      for (int ks = 0; ks < 8; ++ks) {
        v16h b;
        const half_t* q = &cur[(j * 16 + n16) * LDSP + ks * 32 + hf * 8];
        reinterpret_cast<float4*>(&b)[0] = *reinterpret_cast<const float4*>(q);
        reinterpret_cast<float4*>(&b)[1] = *reinterpret_cast<const float4*>(q + 16);
        c = __builtin_amdgcn_wmma_f32_16x16x32_f16(false, af[ks], false, b, (short)0, c, false, false);
      }
      int e = it * 64 + j * 16 + n16;
      float en = enorm[e];
#pragma unroll
      for (int r = 0; r < 8; ++r) {
        float d = vn[r] + en - 2.f * c[r];
        if (d < bd[r]) { bd[r] = d; be[r] = e; }
      }
    }
  }
  // argmin across 16 lanes (first-occurrence tie-break like jnp.argmin)
#pragma unroll
  for (int r = 0; r < 8; ++r) {
#pragma unroll
    for (int m = 1; m <= 8; m <<= 1) {
      float od = __shfl_xor(bd[r], m, 32);
      int oe = __shfl_xor(be[r], m, 32);
      if (od < bd[r] || (od == bd[r] && oe < be[r])) { bd[r] = od; be[r] = oe; }
    }
    if (n16 == 0) idx[rowBase + hf * 8 + r] = be[r];
  }
}

// ---------------- losses ----------------
__global__ __launch_bounds__(256) void k_vqloss(const float* __restrict__ emb, const float* __restrict__ v32,
                                                const int* __restrict__ idx, float* __restrict__ part) {
  __shared__ float red[256];
  float s = 0.f;
  const size_t total = (size_t)NROWS * HH;
  for (size_t i = (size_t)blockIdx.x * 256 + threadIdx.x; i < total; i += (size_t)gridDim.x * 256) {
    size_t row = i >> 8; int n = (int)(i & 255);
    float d = emb[(size_t)idx[row] * HH + n] - v32[i];
    s += d * d;
  }
  red[threadIdx.x] = s; __syncthreads();
  for (int o = 128; o > 0; o >>= 1) { if (threadIdx.x < (unsigned)o) red[threadIdx.x] += red[threadIdx.x + o]; __syncthreads(); }
  if (threadIdx.x == 0) part[blockIdx.x] = red[0];
}

__global__ __launch_bounds__(256) void k_dec(const float* __restrict__ inp, const float* __restrict__ emb,
                                             const int* __restrict__ idx, const float* __restrict__ Wd,
                                             const float* __restrict__ bdv, float* __restrict__ part) {
  __shared__ float q[256];
  __shared__ float red[256];
  int b = blockIdx.x;
  int qi = idx[b * TT + (TT - 1)];
  q[threadIdx.x] = emb[(size_t)qi * HH + threadIdx.x];
  __syncthreads();
  float s = 0.f;
  for (int c = threadIdx.x; c < CC; c += 256) {
    float acc = bdv[c];
    for (int n = 0; n < HH; ++n) acc += q[n] * Wd[(size_t)n * CC + c];
    float y = inp[((size_t)b * CC + c) * TT + (TT - 1)];
    float d = acc - y;
    s += d * d;
  }
  red[threadIdx.x] = s; __syncthreads();
  for (int o = 128; o > 0; o >>= 1) { if (threadIdx.x < (unsigned)o) red[threadIdx.x] += red[threadIdx.x + o]; __syncthreads(); }
  if (threadIdx.x == 0) part[blockIdx.x] = red[0];
}

__global__ __launch_bounds__(256) void k_final(const float* __restrict__ vqpart, const float* __restrict__ decpart,
                                               float* __restrict__ out) {
  __shared__ float red[256];
  __shared__ float vqsum;
  red[threadIdx.x] = vqpart[threadIdx.x] + vqpart[threadIdx.x + 256];
  __syncthreads();
  for (int o = 128; o > 0; o >>= 1) { if (threadIdx.x < (unsigned)o) red[threadIdx.x] += red[threadIdx.x + o]; __syncthreads(); }
  if (threadIdx.x == 0) vqsum = red[0];
  __syncthreads();
  red[threadIdx.x] = (threadIdx.x < 64) ? decpart[threadIdx.x] : 0.f;
  __syncthreads();
  for (int o = 128; o > 0; o >>= 1) { if (threadIdx.x < (unsigned)o) red[threadIdx.x] += red[threadIdx.x + o]; __syncthreads(); }
  if (threadIdx.x == 0)
    out[0] = red[0] / (float)(BB * CC) + 1.02f * vqsum / (float)((size_t)NROWS * HH);
}

// ---------------- launcher ----------------
static inline size_t alup(size_t x) { return (x + 255) & ~(size_t)255; }

extern "C" void kernel_launch(void* const* d_in, const int* in_sizes, int n_in,
                              void* d_out, int out_size, void* d_ws, size_t ws_size,
                              hipStream_t stream) {
  const float* inp   = (const float*)d_in[0];
  const float* W1    = (const float*)d_in[1];
  const float* b1    = (const float*)d_in[2];
  const float* gamma = (const float*)d_in[3];
  const float* beta  = (const float*)d_in[4];
  const float* W2    = (const float*)d_in[5];
  const float* b2    = (const float*)d_in[6];
  const float* emb   = (const float*)d_in[7];
  const float* Wd    = (const float*)d_in[8];
  const float* bdv   = (const float*)d_in[9];
  float* out = (float*)d_out;

  char* ws = (char*)d_ws;
  size_t off = 0;
  half_t* xt   = (half_t*)(ws + off); off = alup(off + (size_t)NROWS * XTS * 2);
  half_t* w1t  = (half_t*)(ws + off); off = alup(off + (size_t)HH * XTS * 2);
  half_t* w2t  = (half_t*)(ws + off); off = alup(off + (size_t)HH * HH * 2);
  half_t* eh   = (half_t*)(ws + off); off = alup(off + (size_t)KK * HH * 2);
  float* enorm = (float*)(ws + off);  off = alup(off + (size_t)KK * 4);
  half_t* vh   = (half_t*)(ws + off); off = alup(off + (size_t)NROWS * HH * 2);
  float* v32   = (float*)(ws + off);  off = alup(off + (size_t)NROWS * HH * 4);
  float* vnorm = (float*)(ws + off);  off = alup(off + (size_t)NROWS * 4);
  int* idx     = (int*)(ws + off);    off = alup(off + (size_t)NROWS * 4);
  float* vqp   = (float*)(ws + off);  off = alup(off + 512 * 4);
  float* decp  = (float*)(ws + off);  off = alup(off + 64 * 4);
  (void)ws_size; (void)in_sizes; (void)n_in; (void)out_size;

  k_xt<<<dim3(9, 17, 64), dim3(32, 8), 0, stream>>>(inp, xt);
  k_w1t<<<256, 256, 0, stream>>>(W1, w1t);
  k_w2t<<<256, 256, 0, stream>>>(W2, w2t);
  k_embed<<<512, 256, 0, stream>>>(emb, eh, enorm);
  k_mlp<<<257, 128, 0, stream>>>(xt, w1t, w2t, b1, gamma, beta, b2, v32, vh, vnorm);
  k_vq<<<257, 128, 0, stream>>>(vh, eh, vnorm, enorm, idx);
  k_vqloss<<<512, 256, 0, stream>>>(emb, v32, idx, vqp);
  k_dec<<<64, 256, 0, stream>>>(inp, emb, idx, Wd, bdv, decp);
  k_final<<<1, 256, 0, stream>>>(vqp, decp, out);
}
